// CapturableTensorNet_76957224010290
// MI455X (gfx1250) — matrix operand order repaired and checked by reading, not compiled, not run on hardware
//
#include <hip/hip_runtime.h>

// ---------------- problem constants (from reference) ----------------
#define HID    128
#define NRBF   32
#define NNODE  4096
#define NEDGE  65536
#define CUTF   4.5f
#define PI_F   3.14159265358979f
#define RBF_ALPHA  (5.0f / 4.5f)
#define RBF_START  0.0111089965f          // exp(-4.5)
#define RBF_STEP   0.0318997099f          // (1-start)/31
#define RBF_BETA   261.7843f              // ((2/32)*(1-start))^-2

typedef __bf16 bf16;
typedef __attribute__((ext_vector_type(16))) __bf16 v16bf;
typedef __attribute__((ext_vector_type(8)))  float  v8f;

union BF16x16 { uint4 u[2]; v16bf v; };

__device__ __forceinline__ bf16 f2bf(float f) { return (bf16)f; }  // native v_cvt
__device__ __forceinline__ float siluf(float x) {
  return x * __builtin_amdgcn_rcpf(1.0f + __expf(-x));
}

// ---------------- WMMA bf16 16x16x32 fragment helpers (wave32) ----------------
// A 16x32 bf16: lanes 0-15 row M=lane, K = {k0..k0+7, k0+16..k0+23};
//               lanes 16-31 row M=lane-16, K = {k0+8..k0+15, k0+24..k0+31}.
__device__ __forceinline__ v16bf frag_a_f32(const float* A, int lda, int k0) {
  int lane = threadIdx.x & 31;
  const float* p = A + (lane & 15) * lda + k0 + ((lane & 16) ? 8 : 0);
  // four independent b128 loads -> single wait, then packed converts
  float4 f0 = *(const float4*)(p);
  float4 f1 = *(const float4*)(p + 4);
  float4 f2 = *(const float4*)(p + 16);
  float4 f3 = *(const float4*)(p + 20);
  v16bf a;
  a[0]  = f2bf(f0.x); a[1]  = f2bf(f0.y); a[2]  = f2bf(f0.z); a[3]  = f2bf(f0.w);
  a[4]  = f2bf(f1.x); a[5]  = f2bf(f1.y); a[6]  = f2bf(f1.z); a[7]  = f2bf(f1.w);
  a[8]  = f2bf(f2.x); a[9]  = f2bf(f2.y); a[10] = f2bf(f2.z); a[11] = f2bf(f2.w);
  a[12] = f2bf(f3.x); a[13] = f2bf(f3.y); a[14] = f2bf(f3.z); a[15] = f2bf(f3.w);
  return a;
}
__device__ __forceinline__ v16bf frag_a_bf(const bf16* A, int lda, int k0) {
  int lane = threadIdx.x & 31;
  const bf16* p = A + (lane & 15) * lda + k0 + ((lane & 16) ? 8 : 0);
  BF16x16 cv;
  cv.u[0] = *(const uint4*)(p);        // K  +0..7
  cv.u[1] = *(const uint4*)(p + 16);   // K +16..23
  return cv.v;
}
// B 32x16 bf16 from row-major weight W[Nout,K] (so B = W^T):
// lanes 0-15 col N=lane K={k0..k0+7, k0+16..k0+23}; lanes 16-31 col N=lane-16 K={k0+8..k0+15, k0+24..k0+31}.
__device__ __forceinline__ v16bf frag_b_w(const bf16* W, int K, int n0, int k0) {
  int lane = threadIdx.x & 31;
  const bf16* p = W + (n0 + (lane & 15)) * K + k0 + ((lane & 16) ? 8 : 0);
  BF16x16 cv;
  cv.u[0] = *(const uint4*)(p);
  cv.u[1] = *(const uint4*)(p + 16);
  return cv.v;
}
__device__ __forceinline__ v8f wmma_bf(v16bf a, v16bf b, v8f c) {
  return __builtin_amdgcn_wmma_f32_16x16x32_bf16(false, a, false, b, (short)0, c,
                                                 false, false);
}

// ---------------- 3x3 compact-tensor helpers ----------------
// compact form: it (trace coeff), a[3] (skew: A=[[0,-a2,a1],[a2,0,-a0],[-a1,a0,0]]),
// s[6] (sym traceless: xx,xy,xz,yy,yz,zz)
__device__ __forceinline__ void decomp9(const float* x, float& it, float* a, float* s) {
  it = (x[0] + x[4] + x[8]) * (1.0f / 3.0f);
  a[0] = 0.5f * (x[7] - x[5]);
  a[1] = 0.5f * (x[2] - x[6]);
  a[2] = 0.5f * (x[3] - x[1]);
  s[0] = x[0] - it;
  s[1] = 0.5f * (x[1] + x[3]);
  s[2] = 0.5f * (x[2] + x[6]);
  s[3] = x[4] - it;
  s[4] = 0.5f * (x[5] + x[7]);
  s[5] = x[8] - it;
}
__device__ __forceinline__ void full9(float it, const float* a, const float* s, float* m) {
  m[0] = it + s[0]; m[1] = s[1] - a[2]; m[2] = s[2] + a[1];
  m[3] = s[1] + a[2]; m[4] = it + s[3]; m[5] = s[4] - a[0];
  m[6] = s[2] - a[1]; m[7] = s[4] + a[0]; m[8] = it + s[5];
}

// ---------------- tiny utility kernels ----------------
__global__ void tn_cvt(const float* __restrict__ s, bf16* __restrict__ d, int n) {
  int i = blockIdx.x * blockDim.x + threadIdx.x;
  if (i < n) d[i] = f2bf(s[i]);
}
__global__ void tn_copy(const float* __restrict__ s, float* __restrict__ d, int n) {
  int i = blockIdx.x * blockDim.x + threadIdx.x;
  if (i < n) d[i] = s[i];
}
__global__ void tn_zero(float* __restrict__ p, int n) {
  int i = blockIdx.x * blockDim.x + threadIdx.x;
  if (i < n) p[i] = 0.0f;
}

// -------- generic WMMA GEMM: C[M,Nout] = act(A[M,K] @ W[Nout,K]^T + b) ----------
// one wave computes a 16x64 tile (A fragment reused across 4 WMMAs per k-step)
__global__ __launch_bounds__(128) void tn_gemm(
    const float* __restrict__ A, int lda,
    const bf16* __restrict__ W, const float* __restrict__ bias,
    float* __restrict__ C, int ldc, int M, int Nout, int K, int act) {
  int wave = blockIdx.x * 4 + (threadIdx.x >> 5);
  int tn = Nout >> 6;
  if (wave >= (M >> 4) * tn) return;
  int m0 = (wave / tn) << 4;
  int n0 = (wave % tn) << 6;
  v8f acc0 = {}, acc1 = {}, acc2 = {}, acc3 = {};
  for (int k0 = 0; k0 < K; k0 += 32) {
    v16bf a = frag_a_f32(A + m0 * lda, lda, k0);
    acc0 = wmma_bf(a, frag_b_w(W, K, n0, k0), acc0);
    acc1 = wmma_bf(a, frag_b_w(W, K, n0 + 16, k0), acc1);
    acc2 = wmma_bf(a, frag_b_w(W, K, n0 + 32, k0), acc2);
    acc3 = wmma_bf(a, frag_b_w(W, K, n0 + 48, k0), acc3);
  }
  int lane = threadIdx.x & 31;
  int nn = lane & 15;
  int mo = (lane & 16) ? 8 : 0;
  v8f accs[4] = {acc0, acc1, acc2, acc3};
#pragma unroll
  for (int q = 0; q < 4; ++q) {
    int nc = n0 + q * 16 + nn;
    float bs = bias ? bias[nc] : 0.0f;
#pragma unroll
    for (int r = 0; r < 8; ++r) {
      float v = accs[q][r] + bs;
      if (act) v = siluf(v);
      C[(m0 + mo + r) * ldc + nc] = v;
    }
  }
}

// ---------------- fused embedding edge kernel (1 wave = 16 edges) ----------------
__global__ __launch_bounds__(32) void tn_embed(
    const int* __restrict__ z, const int* __restrict__ ei,
    const float* __restrict__ ew, const float* __restrict__ evn,
    const float* __restrict__ emb,
    const bf16* __restrict__ w_emb2, const float* __restrict__ b_emb2,
    const bf16* __restrict__ w_dpc, const float* __restrict__ b_dpc,
    bf16* __restrict__ eaBF, float* __restrict__ Cbuf,
    float* __restrict__ It, float* __restrict__ Av, float* __restrict__ Sm) {
  __shared__ alignas(16) bf16  sZ[16 * 256];    // [Z[dst] | Z[src]] bf16
  __shared__ alignas(16) bf16  sEA[16 * 32];
  __shared__ alignas(16) float sZij[16 * 128];
  __shared__ alignas(16) float sW[16 * 384];    // [W1|W2|W3]
  __shared__ float sC[16];
  __shared__ float sV[16][3];
  __shared__ int   sSrc[16], sDst[16];

  const int lane = threadIdx.x;
  const int e0 = blockIdx.x * 16;
  if (lane == 0) {
    __builtin_prefetch(w_emb2, 0, 3);
    __builtin_prefetch(w_dpc, 0, 3);
  }

  if (lane < 16) {
    int e = e0 + lane;
    float d = ew[e];
    float cf = (d < CUTF) ? 0.5f * (__cosf(d * (PI_F / CUTF)) + 1.0f) : 0.0f;
    sC[lane] = cf;
    Cbuf[e] = cf;
    sV[lane][0] = evn[e * 3 + 0];
    sV[lane][1] = evn[e * 3 + 1];
    sV[lane][2] = evn[e * 3 + 2];
    sSrc[lane] = ei[e];
    sDst[lane] = ei[NEDGE + e];
    float ex = __expf(-RBF_ALPHA * d);
#pragma unroll
    for (int k = 0; k < NRBF; ++k) {
      float t = ex - (RBF_START + (float)k * RBF_STEP);
      bf16 v = f2bf(cf * __expf(-RBF_BETA * t * t));
      sEA[lane * 32 + k] = v;
      eaBF[e * 32 + k] = v;
    }
  }
  __syncthreads();
  for (int idx = lane; idx < 16 * 256; idx += 32) {
    int e = idx >> 8, c = idx & 255;
    int node = (c < 128) ? sDst[e] : sSrc[e];
    sZ[idx] = f2bf(emb[z[node] * HID + (c & 127)]);
  }
  __syncthreads();
  const int nn = lane & 15;
  const int mo = (lane & 16) ? 8 : 0;
  // hoist A fragments (register-resident across all column tiles)
  v16bf az[8];
#pragma unroll
  for (int kt = 0; kt < 8; ++kt) az[kt] = frag_a_bf(sZ, 256, kt * 32);
  v16bf ae = frag_a_bf(sEA, 32, 0);
  // Zij[16,128] = Zcat @ emb2_w^T + b   (K=256)
  for (int nt = 0; nt < 8; ++nt) {
    v8f acc = {};
#pragma unroll
    for (int kt = 0; kt < 8; ++kt)
      acc = wmma_bf(az[kt], frag_b_w(w_emb2, 256, nt * 16, kt * 32), acc);
    float bs = b_emb2[nt * 16 + nn];
#pragma unroll
    for (int r = 0; r < 8; ++r) sZij[(mo + r) * 128 + nt * 16 + nn] = acc[r] + bs;
  }
  // W123[16,384] = (ea @ dpcat^T + b) * C    (K=32)
  for (int nt = 0; nt < 24; ++nt) {
    v8f acc = wmma_bf(ae, frag_b_w(w_dpc, 32, nt * 16, 0), (v8f){});
    float bs = b_dpc[nt * 16 + nn];
#pragma unroll
    for (int r = 0; r < 8; ++r)
      sW[(mo + r) * 384 + nt * 16 + nn] = (acc[r] + bs) * sC[mo + r];
  }
  __syncthreads();
  // scatter: segment-sum over dst into compact node tensors
  for (int e = 0; e < 16; ++e) {
    int dn = sDst[e];
    float vx = sV[e][0], vy = sV[e][1], vz = sV[e][2];
    float sy0 = vx * vx - (1.0f / 3.0f), sy1 = vx * vy, sy2 = vx * vz;
    float sy3 = vy * vy - (1.0f / 3.0f), sy4 = vy * vz, sy5 = vz * vz - (1.0f / 3.0f);
#pragma unroll
    for (int t = 0; t < 4; ++t) {
      int h = lane + t * 32;
      float zij = sZij[e * 128 + h];
      float s1 = zij * sW[e * 384 + h];
      float s2 = zij * sW[e * 384 + 128 + h];
      float s3 = zij * sW[e * 384 + 256 + h];
      atomicAdd(&It[dn * HID + h], s1);
      atomicAdd(&Av[(dn * 3 + 0) * HID + h], s2 * vx);
      atomicAdd(&Av[(dn * 3 + 1) * HID + h], s2 * vy);
      atomicAdd(&Av[(dn * 3 + 2) * HID + h], s2 * vz);
      atomicAdd(&Sm[(dn * 6 + 0) * HID + h], s3 * sy0);
      atomicAdd(&Sm[(dn * 6 + 1) * HID + h], s3 * sy1);
      atomicAdd(&Sm[(dn * 6 + 2) * HID + h], s3 * sy2);
      atomicAdd(&Sm[(dn * 6 + 3) * HID + h], s3 * sy3);
      atomicAdd(&Sm[(dn * 6 + 4) * HID + h], s3 * sy4);
      atomicAdd(&Sm[(dn * 6 + 5) * HID + h], s3 * sy5);
    }
  }
}

// ---------------- fused per-layer message kernel (1 wave = 16 edges) -------------
__global__ __launch_bounds__(32) void tn_msg(
    const bf16* __restrict__ eaBF, const float* __restrict__ Cbuf,
    const int* __restrict__ ei,
    const bf16* __restrict__ w1, const float* __restrict__ b1,
    const bf16* __restrict__ w2, const float* __restrict__ b2,
    const bf16* __restrict__ w3, const float* __restrict__ b3,
    const float* __restrict__ Yit, const float* __restrict__ Yav,
    const float* __restrict__ Ysm,
    float* __restrict__ Mit, float* __restrict__ Mav, float* __restrict__ Msm) {
  __shared__ alignas(16) bf16  sEA[16 * 32];
  __shared__ alignas(16) float sT1[16 * 128];
  __shared__ alignas(16) float sT2[16 * 256];
  __shared__ alignas(16) float sAt[16 * 384];
  __shared__ float sC[16];
  __shared__ int   sSrc[16], sDst[16];
  const int lane = threadIdx.x;
  const int e0 = blockIdx.x * 16;
  if (lane == 0) {
    __builtin_prefetch(w1, 0, 3);
    __builtin_prefetch(w2, 0, 3);
    __builtin_prefetch(w3, 0, 3);
  }
  for (int idx = lane; idx < 16 * 32; idx += 32) sEA[idx] = eaBF[e0 * 32 + idx];
  if (lane < 16) {
    int e = e0 + lane;
    sC[lane] = Cbuf[e];
    sSrc[lane] = ei[e];
    sDst[lane] = ei[NEDGE + e];
  }
  __syncthreads();
  const int nn = lane & 15;
  const int mo = (lane & 16) ? 8 : 0;
  v16bf ae = frag_a_bf(sEA, 32, 0);
  for (int nt = 0; nt < 8; ++nt) {                 // 32 -> 128, silu
    v8f acc = wmma_bf(ae, frag_b_w(w1, 32, nt * 16, 0), (v8f){});
    float bs = b1[nt * 16 + nn];
#pragma unroll
    for (int r = 0; r < 8; ++r) sT1[(mo + r) * 128 + nt * 16 + nn] = siluf(acc[r] + bs);
  }
  __syncthreads();
  v16bf a1[4];
#pragma unroll
  for (int kt = 0; kt < 4; ++kt) a1[kt] = frag_a_f32(sT1, 128, kt * 32);
  for (int nt = 0; nt < 16; ++nt) {                // 128 -> 256, silu
    v8f acc = {};
#pragma unroll
    for (int kt = 0; kt < 4; ++kt)
      acc = wmma_bf(a1[kt], frag_b_w(w2, 128, nt * 16, kt * 32), acc);
    float bs = b2[nt * 16 + nn];
#pragma unroll
    for (int r = 0; r < 8; ++r) sT2[(mo + r) * 256 + nt * 16 + nn] = siluf(acc[r] + bs);
  }
  __syncthreads();
  v16bf a2[8];
#pragma unroll
  for (int kt = 0; kt < 8; ++kt) a2[kt] = frag_a_f32(sT2, 256, kt * 32);
  for (int nt = 0; nt < 24; ++nt) {                // 256 -> 384, silu, *C
    v8f acc = {};
#pragma unroll
    for (int kt = 0; kt < 8; ++kt)
      acc = wmma_bf(a2[kt], frag_b_w(w3, 256, nt * 16, kt * 32), acc);
    float bs = b3[nt * 16 + nn];
#pragma unroll
    for (int r = 0; r < 8; ++r)
      sAt[(mo + r) * 384 + nt * 16 + nn] = siluf(acc[r] + bs) * sC[mo + r];
  }
  __syncthreads();
  for (int e = 0; e < 16; ++e) {
    int sn = sSrc[e], dn = sDst[e];
#pragma unroll
    for (int t = 0; t < 4; ++t) {
      int h = lane + t * 32;
      float a0 = sAt[e * 384 + h * 3 + 0];
      float av = sAt[e * 384 + h * 3 + 1];
      float as = sAt[e * 384 + h * 3 + 2];
      atomicAdd(&Mit[dn * HID + h], a0 * Yit[sn * HID + h]);
#pragma unroll
      for (int c = 0; c < 3; ++c)
        atomicAdd(&Mav[(dn * 3 + c) * HID + h], av * Yav[(sn * 3 + c) * HID + h]);
#pragma unroll
      for (int c = 0; c < 6; ++c)
        atomicAdd(&Msm[(dn * 6 + c) * HID + h], as * Ysm[(sn * 6 + c) * HID + h]);
    }
  }
}

// ---------------- element-wise node kernels ----------------
__global__ void tn_tnorm0(const float* __restrict__ It, const float* __restrict__ Av,
                          const float* __restrict__ Sm, float* __restrict__ tb) {
  int i = blockIdx.x * blockDim.x + threadIdx.x;
  if (i >= NNODE * HID) return;
  int n = i / HID, h = i % HID;
  float it = It[i];
  float a0 = Av[(n * 3 + 0) * HID + h], a1 = Av[(n * 3 + 1) * HID + h],
        a2 = Av[(n * 3 + 2) * HID + h];
  float s0 = Sm[(n * 6 + 0) * HID + h], s1 = Sm[(n * 6 + 1) * HID + h],
        s2 = Sm[(n * 6 + 2) * HID + h], s3 = Sm[(n * 6 + 3) * HID + h],
        s4 = Sm[(n * 6 + 4) * HID + h], s5 = Sm[(n * 6 + 5) * HID + h];
  tb[i] = 3.0f * it * it + 2.0f * (a0 * a0 + a1 * a1 + a2 * a2) +
          s0 * s0 + s3 * s3 + s5 * s5 + 2.0f * (s1 * s1 + s2 * s2 + s4 * s4);
}

__global__ __launch_bounds__(128) void tn_ln(const float* __restrict__ in,
                                             const float* __restrict__ w,
                                             const float* __restrict__ b,
                                             float* __restrict__ out, int cols) {
  __shared__ float red[128];
  int row = blockIdx.x, t = threadIdx.x;
  const float* p = in + row * cols;
  float s = 0.0f, s2 = 0.0f;
  for (int c = t; c < cols; c += 128) {
    float v = p[c];
    s += v;
    s2 += v * v;
  }
  red[t] = s;
  __syncthreads();
  for (int off = 64; off > 0; off >>= 1) {
    if (t < off) red[t] += red[t + off];
    __syncthreads();
  }
  float mean = red[0] / (float)cols;
  __syncthreads();
  red[t] = s2;
  __syncthreads();
  for (int off = 64; off > 0; off >>= 1) {
    if (t < off) red[t] += red[t + off];
    __syncthreads();
  }
  float var = red[0] / (float)cols - mean * mean;
  float inv = rsqrtf(var + 1e-5f);
  for (int c = t; c < cols; c += 128)
    out[row * cols + c] = (p[c] - mean) * inv * w[c] + b[c];
}

__global__ void tn_assembleX(const float* __restrict__ Il, const float* __restrict__ Al,
                             const float* __restrict__ Sl, const float* __restrict__ nvec,
                             float* __restrict__ X) {
  int i = blockIdx.x * blockDim.x + threadIdx.x;
  if (i >= NNODE * HID) return;
  int n = i / HID, h = i % HID;
  float n0 = nvec[n * 384 + h * 3 + 0];
  float n1 = nvec[n * 384 + h * 3 + 1];
  float n2 = nvec[n * 384 + h * 3 + 2];
  float it = n0 * Il[i];
  float a[3], s[6], m[9];
#pragma unroll
  for (int c = 0; c < 3; ++c) a[c] = n1 * Al[(n * 3 + c) * HID + h];
#pragma unroll
  for (int c = 0; c < 6; ++c) s[c] = n2 * Sl[(n * 6 + c) * HID + h];
  full9(it, a, s, m);
  float* xp = X + (size_t)i * 9;
#pragma unroll
  for (int j = 0; j < 9; ++j) xp[j] = m[j];
}

__global__ void tn_prep(float* __restrict__ X, float* __restrict__ It,
                        float* __restrict__ Av, float* __restrict__ Sm) {
  int i = blockIdx.x * blockDim.x + threadIdx.x;
  if (i >= NNODE * HID) return;
  int n = i / HID, h = i % HID;
  float* xp = X + (size_t)i * 9;
  float x[9], tn = 0.0f;
#pragma unroll
  for (int j = 0; j < 9; ++j) {
    x[j] = xp[j];
    tn += x[j] * x[j];
  }
  float sc = 1.0f / (tn + 1.0f);
#pragma unroll
  for (int j = 0; j < 9; ++j) {
    x[j] *= sc;
    xp[j] = x[j];
  }
  float it, a[3], s[6];
  decomp9(x, it, a, s);
  It[i] = it;
#pragma unroll
  for (int c = 0; c < 3; ++c) Av[(n * 3 + c) * HID + h] = a[c];
#pragma unroll
  for (int c = 0; c < 6; ++c) Sm[(n * 6 + c) * HID + h] = s[c];
}

__global__ void tn_update(const float* __restrict__ Mit, const float* __restrict__ Mav,
                          const float* __restrict__ Msm, const float* __restrict__ Yit,
                          const float* __restrict__ Yav, const float* __restrict__ Ysm,
                          float* __restrict__ It, float* __restrict__ Av,
                          float* __restrict__ Sm) {
  int i = blockIdx.x * blockDim.x + threadIdx.x;
  if (i >= NNODE * HID) return;
  int n = i / HID, h = i % HID;
  float ma[3], ms[6], ya[3], ys[6], Mf[9], Yf[9], P[9];
#pragma unroll
  for (int c = 0; c < 3; ++c) {
    ma[c] = Mav[(n * 3 + c) * HID + h];
    ya[c] = Yav[(n * 3 + c) * HID + h];
  }
#pragma unroll
  for (int c = 0; c < 6; ++c) {
    ms[c] = Msm[(n * 6 + c) * HID + h];
    ys[c] = Ysm[(n * 6 + c) * HID + h];
  }
  full9(Mit[i], ma, ms, Mf);
  full9(Yit[i], ya, ys, Yf);
  float tn = 0.0f;
#pragma unroll
  for (int r = 0; r < 3; ++r)
#pragma unroll
    for (int c = 0; c < 3; ++c) {
      float acc = 0.0f;
#pragma unroll
      for (int k = 0; k < 3; ++k)
        acc += Mf[r * 3 + k] * Yf[k * 3 + c] + Yf[r * 3 + k] * Mf[k * 3 + c];
      P[r * 3 + c] = acc;
      tn += acc * acc;
    }
  float inv = 1.0f / (tn + 1.0f);
  float it, a[3], s[6];
  decomp9(P, it, a, s);
  It[i] = it * inv;
#pragma unroll
  for (int c = 0; c < 3; ++c) Av[(n * 3 + c) * HID + h] = a[c] * inv;
#pragma unroll
  for (int c = 0; c < 6; ++c) Sm[(n * 6 + c) * HID + h] = s[c] * inv;
}

__global__ void tn_xupd(const float* __restrict__ Il, const float* __restrict__ Al,
                        const float* __restrict__ Sl, float* __restrict__ X) {
  int i = blockIdx.x * blockDim.x + threadIdx.x;
  if (i >= NNODE * HID) return;
  int n = i / HID, h = i % HID;
  float a[3], s[6], d[9];
#pragma unroll
  for (int c = 0; c < 3; ++c) a[c] = Al[(n * 3 + c) * HID + h];
#pragma unroll
  for (int c = 0; c < 6; ++c) s[c] = Sl[(n * 6 + c) * HID + h];
  full9(Il[i], a, s, d);
  float* xp = X + (size_t)i * 9;
#pragma unroll
  for (int r = 0; r < 3; ++r)
#pragma unroll
    for (int c = 0; c < 3; ++c) {
      float d2 = 0.0f;
#pragma unroll
      for (int k = 0; k < 3; ++k) d2 += d[r * 3 + k] * d[k * 3 + c];
      xp[r * 3 + c] += d[r * 3 + c] + d2;
    }
}

__global__ void tn_head(const float* __restrict__ X, float* __restrict__ xcat) {
  int i = blockIdx.x * blockDim.x + threadIdx.x;
  if (i >= NNODE * HID) return;
  int n = i / HID, h = i % HID;
  const float* xp = X + (size_t)i * 9;
  float x[9];
#pragma unroll
  for (int j = 0; j < 9; ++j) x[j] = xp[j];
  float it, a[3], s[6];
  decomp9(x, it, a, s);
  xcat[n * 384 + h]       = 3.0f * it * it;
  xcat[n * 384 + 128 + h] = 2.0f * (a[0] * a[0] + a[1] * a[1] + a[2] * a[2]);
  xcat[n * 384 + 256 + h] = s[0] * s[0] + s[3] * s[3] + s[5] * s[5] +
                            2.0f * (s[1] * s[1] + s[2] * s[2] + s[4] * s[4]);
}

// ---------------- host orchestration ----------------
extern "C" void kernel_launch(void* const* d_in, const int* in_sizes, int n_in,
                              void* d_out, int out_size, void* d_ws, size_t ws_size,
                              hipStream_t stream) {
  (void)in_sizes; (void)n_in; (void)out_size; (void)ws_size;
  const int*   z     = (const int*)d_in[0];
  const int*   ei    = (const int*)d_in[1];
  const float* ew    = (const float*)d_in[2];
  const float* evn   = (const float*)d_in[3];
  const float* emb   = (const float*)d_in[4];
  const float* dp1w  = (const float*)d_in[5];
  const float* dp1b  = (const float*)d_in[6];
  const float* dp2w  = (const float*)d_in[7];
  const float* dp2b  = (const float*)d_in[8];
  const float* dp3w  = (const float*)d_in[9];
  const float* dp3b  = (const float*)d_in[10];
  const float* emb2w = (const float*)d_in[11];
  const float* emb2b = (const float*)d_in[12];
  const float* tls1w = (const float*)d_in[13];
  const float* tls1b = (const float*)d_in[14];
  const float* tls2w = (const float*)d_in[15];
  const float* tls2b = (const float*)d_in[16];
  const float* outw  = (const float*)d_in[17];
  const float* outb  = (const float*)d_in[18];
  const float* telt[3] = {(const float*)d_in[19], (const float*)d_in[20],
                          (const float*)d_in[21]};
  const float* initw = (const float*)d_in[22];
  const float* initb = (const float*)d_in[23];
  const float* outnw = (const float*)d_in[24];
  const float* outnb = (const float*)d_in[25];
  struct LayerIn { const float *ls1w, *ls1b, *ls2w, *ls2b, *ls3w, *ls3b, *lt[6]; } L[2];
  for (int l = 0; l < 2; ++l) {
    int base = 26 + l * 12;
    L[l].ls1w = (const float*)d_in[base + 0];
    L[l].ls1b = (const float*)d_in[base + 1];
    L[l].ls2w = (const float*)d_in[base + 2];
    L[l].ls2b = (const float*)d_in[base + 3];
    L[l].ls3w = (const float*)d_in[base + 4];
    L[l].ls3b = (const float*)d_in[base + 5];
    for (int j = 0; j < 6; ++j) L[l].lt[j] = (const float*)d_in[base + 6 + j];
  }

  char* cur = (char*)d_ws;
  auto alloc = [&](size_t bytes) -> void* {
    void* p = (void*)cur;
    cur += (bytes + 255) & ~(size_t)255;
    return p;
  };
  bf16*  wDPC  = (bf16*)alloc(384 * 32 * 2);
  float* bDPC  = (float*)alloc(384 * 4);
  bf16*  wEMB2 = (bf16*)alloc(128 * 256 * 2);
  bf16*  wTLS1 = (bf16*)alloc(256 * 128 * 2);
  bf16*  wTLS2 = (bf16*)alloc(384 * 256 * 2);
  bf16*  wTLT[3];
  for (int j = 0; j < 3; ++j) wTLT[j] = (bf16*)alloc(128 * 128 * 2);
  bf16* wOUT = (bf16*)alloc(128 * 384 * 2);
  struct LayerW { bf16 *ls1, *ls2, *ls3, *lt[6]; } LW[2];
  for (int l = 0; l < 2; ++l) {
    LW[l].ls1 = (bf16*)alloc(128 * 32 * 2);
    LW[l].ls2 = (bf16*)alloc(256 * 128 * 2);
    LW[l].ls3 = (bf16*)alloc(384 * 256 * 2);
    for (int j = 0; j < 6; ++j) LW[l].lt[j] = (bf16*)alloc(128 * 128 * 2);
  }
  bf16*  eaBF = (bf16*)alloc((size_t)NEDGE * 32 * 2);
  float* Cbuf = (float*)alloc((size_t)NEDGE * 4);
  float* It   = (float*)alloc((size_t)NNODE * HID * 4);
  float* Av   = (float*)alloc((size_t)NNODE * 3 * HID * 4);
  float* Sm   = (float*)alloc((size_t)NNODE * 6 * HID * 4);
  float* Il   = (float*)alloc((size_t)NNODE * HID * 4);
  float* Al   = (float*)alloc((size_t)NNODE * 3 * HID * 4);
  float* Sl   = (float*)alloc((size_t)NNODE * 6 * HID * 4);
  float* Mit  = (float*)alloc((size_t)NNODE * HID * 4);
  float* Mav  = (float*)alloc((size_t)NNODE * 3 * HID * 4);
  float* Msm  = (float*)alloc((size_t)NNODE * 6 * HID * 4);
  float* X    = (float*)alloc((size_t)NNODE * HID * 9 * 4);
  float* tbuf = (float*)alloc((size_t)NNODE * HID * 4);
  float* nrm  = (float*)alloc((size_t)NNODE * HID * 4);
  float* mlp1 = (float*)alloc((size_t)NNODE * 256 * 4);
  float* nvec = (float*)alloc((size_t)NNODE * 384 * 4);
  float* xcat = (float*)alloc((size_t)NNODE * 384 * 4);
  float* xln  = (float*)alloc((size_t)NNODE * 384 * 4);

  auto cvt = [&](const float* s, bf16* d, int n) {
    tn_cvt<<<(n + 255) / 256, 256, 0, stream>>>(s, d, n);
  };
  auto cpy = [&](const float* s, float* d, int n) {
    tn_copy<<<(n + 255) / 256, 256, 0, stream>>>(s, d, n);
  };
  auto zero = [&](float* p, int n) {
    tn_zero<<<(n + 255) / 256, 256, 0, stream>>>(p, n);
  };
  auto gemm = [&](const float* A, int lda, const bf16* W, const float* bias, float* Cm,
                  int ldc, int M, int Nout, int K, int act) {
    int tiles = (M / 16) * (Nout / 64);
    tn_gemm<<<(tiles + 3) / 4, 128, 0, stream>>>(A, lda, W, bias, Cm, ldc, M, Nout, K, act);
  };

  // weight conversion / packing
  cvt(dp1w, wDPC, 128 * 32);
  cvt(dp2w, wDPC + 128 * 32, 128 * 32);
  cvt(dp3w, wDPC + 256 * 32, 128 * 32);
  cpy(dp1b, bDPC, 128);
  cpy(dp2b, bDPC + 128, 128);
  cpy(dp3b, bDPC + 256, 128);
  cvt(emb2w, wEMB2, 128 * 256);
  cvt(tls1w, wTLS1, 256 * 128);
  cvt(tls2w, wTLS2, 384 * 256);
  for (int j = 0; j < 3; ++j) cvt(telt[j], wTLT[j], 128 * 128);
  cvt(outw, wOUT, 128 * 384);
  for (int l = 0; l < 2; ++l) {
    cvt(L[l].ls1w, LW[l].ls1, 128 * 32);
    cvt(L[l].ls2w, LW[l].ls2, 256 * 128);
    cvt(L[l].ls3w, LW[l].ls3, 384 * 256);
    for (int j = 0; j < 6; ++j) cvt(L[l].lt[j], LW[l].lt[j], 128 * 128);
  }

  const int NH = NNODE * HID;
  const int EB = (NH + 255) / 256;

  // ---- TensorEmbedding ----
  zero(It, NH);
  zero(Av, NH * 3);
  zero(Sm, NH * 6);
  tn_embed<<<NEDGE / 16, 32, 0, stream>>>(z, ei, ew, evn, emb, wEMB2, emb2b, wDPC, bDPC,
                                          eaBF, Cbuf, It, Av, Sm);
  tn_tnorm0<<<EB, 256, 0, stream>>>(It, Av, Sm, tbuf);
  tn_ln<<<NNODE, 128, 0, stream>>>(tbuf, initw, initb, nrm, 128);
  gemm(nrm, 128, wTLS1, tls1b, mlp1, 256, NNODE, 256, 128, 1);
  gemm(mlp1, 256, wTLS2, tls2b, nvec, 384, NNODE, 384, 256, 1);
  gemm(It, 128, wTLT[0], nullptr, Il, 128, NNODE, 128, 128, 0);
  gemm(Av, 128, wTLT[1], nullptr, Al, 128, NNODE * 3, 128, 128, 0);
  gemm(Sm, 128, wTLT[2], nullptr, Sl, 128, NNODE * 6, 128, 128, 0);
  tn_assembleX<<<EB, 256, 0, stream>>>(Il, Al, Sl, nvec, X);

  // ---- interaction layers ----
  for (int l = 0; l < 2; ++l) {
    tn_prep<<<EB, 256, 0, stream>>>(X, It, Av, Sm);
    gemm(It, 128, LW[l].lt[0], nullptr, Il, 128, NNODE, 128, 128, 0);
    gemm(Av, 128, LW[l].lt[1], nullptr, Al, 128, NNODE * 3, 128, 128, 0);
    gemm(Sm, 128, LW[l].lt[2], nullptr, Sl, 128, NNODE * 6, 128, 128, 0);
    zero(Mit, NH);
    zero(Mav, NH * 3);
    zero(Msm, NH * 6);
    tn_msg<<<NEDGE / 16, 32, 0, stream>>>(eaBF, Cbuf, ei, LW[l].ls1, L[l].ls1b, LW[l].ls2,
                                          L[l].ls2b, LW[l].ls3, L[l].ls3b, Il, Al, Sl,
                                          Mit, Mav, Msm);
    tn_update<<<EB, 256, 0, stream>>>(Mit, Mav, Msm, Il, Al, Sl, It, Av, Sm);
    gemm(It, 128, LW[l].lt[3], nullptr, Il, 128, NNODE, 128, 128, 0);
    gemm(Av, 128, LW[l].lt[4], nullptr, Al, 128, NNODE * 3, 128, 128, 0);
    gemm(Sm, 128, LW[l].lt[5], nullptr, Sl, 128, NNODE * 6, 128, 128, 0);
    tn_xupd<<<EB, 256, 0, stream>>>(Il, Al, Sl, X);
  }

  // ---- output head ----
  tn_head<<<EB, 256, 0, stream>>>(X, xcat);
  tn_ln<<<NNODE, 128, 0, stream>>>(xcat, outnw, outnb, xln, 384);
  gemm(xln, 384, wOUT, outb, (float*)d_out, 128, NNODE, 128, 384, 1);
}
